// LSTM_81535659147893
// MI455X (gfx1250) — matrix operand order repaired and checked by reading, not compiled, not run on hardware
//
#include <hip/hip_runtime.h>
#include <hip/hip_bf16.h>

// ---------------------------------------------------------------------------
// LSTM (B=64, T=512, D=512) for gfx1250 using v_wmma_f32_16x16x32_f16.
//  Phase 0: f32 -> f16 conversion of X and the 7 used weight matrices.
//  Phase 1: input projections Xi/Xc/Xo = X*W^T + b  (Wf/Xf is dead code in
//           the reference), stored [T,B,D] f16 so each timestep is contiguous.
//  Phase 2: 512 sequential step kernels. Each block (4 waves) owns one
//           16-column n-tile for ALL four gates; the shared U slab for each
//           32-wide K chunk is staged into double-buffered LDS with
//           global_load_async_to_lds_b128 (ASYNCcnt), overlapping the next
//           chunk's copy with the current chunk's WMMAs, then consumed via
//           bank-conflict-free ds_load_b128 B-fragment reads.
// ---------------------------------------------------------------------------

typedef __attribute__((ext_vector_type(16))) _Float16 v16h;
typedef __attribute__((ext_vector_type(8)))  _Float16 v8h;
typedef __attribute__((ext_vector_type(4)))  _Float16 v4h;
typedef __attribute__((ext_vector_type(8)))  float    v8f;
typedef __attribute__((ext_vector_type(4)))  float    v4f;

#define B_   64
#define T_   512
#define D_   512
#define M_   (B_ * T_)                    // 32768 rows for the projection GEMM
#define TBD  ((size_t)T_ * B_ * D_)       // elements of one pre-activation tensor

// LDS staging geometry for the recurrence (per 32-wide K chunk):
//   4 gates x 16 rows x 64B of K data, row stride padded 64->80B so the 16
//   b128 lanes of a B-fragment read land on distinct banks (20*lr mod 64 is
//   a permutation), and 16B alignment is preserved.
#define ROWSTRIDE_   80
#define GSLAB_       (16 * ROWSTRIDE_)    // 1280 B per gate
#define CHUNK_BYTES_ (4 * GSLAB_)         // 5120 B per K chunk
#define NKCH_        (D_ / 32)            // 16 K chunks

union V16U { v16h v; v8h h[2]; };

// A-fragment (16x32 f16, MxK): lane = 16*hi + lr holds row M=lr,
// K = {hi*16 .. hi*16+15} contiguous (matching the B fragments).
__device__ __forceinline__ v16h load_a_frag(const _Float16* __restrict__ base,
                                            int row, int k0, int hi, int ldk) {
  const _Float16* p = base + (size_t)row * ldk + k0 + hi * 16;
  V16U u;
  u.h[0] = *(const v8h*)(p);
  u.h[1] = *(const v8h*)(p + 8);
  return u.v;
}

// B-fragment (32x16 f16, KxN): lane = 16*hi + lr holds column N=lr,
// K = {hi*16 .. hi*16+15} contiguous (the k order this code uses for both A and B).
// Our B column n is W row n (Y = X * W^T), so this is a contiguous load.
__device__ __forceinline__ v16h load_b_frag(const _Float16* __restrict__ base,
                                            int row, int k0, int hi, int ldk) {
  const _Float16* p = base + (size_t)row * ldk + k0 + hi * 16;
  V16U u;
  u.h[0] = *(const v8h*)(p);
  u.h[1] = *(const v8h*)(p + 8);
  return u.v;
}

// B-fragment from the padded LDS slab (two ds_load_b128 per lane).
__device__ __forceinline__ v16h lds_b_frag(const char* slab, int g, int lr, int hi) {
  const v8h* p = (const v8h*)(slab + g * GSLAB_ + lr * ROWSTRIDE_ + hi * 32);
  V16U u;
  u.h[0] = p[0];
  u.h[1] = p[1];
  return u.v;
}

__device__ __forceinline__ v8f wmma16(v16h a, v16h b, v8f c) {
  return __builtin_amdgcn_wmma_f32_16x16x32_f16(false, a, false, b,
                                                (short)0, c, false, false);
}

__device__ __forceinline__ float sigm_(float x) {
  x = fminf(15.0f, fmaxf(-15.0f, x));
  return 1.0f / (1.0f + __expf(-x));
}
__device__ __forceinline__ float tanh_(float x) {
  x = fminf(15.0f, fmaxf(-15.0f, x));
  float e = __expf(2.0f * x);
  return (e - 1.0f) / (e + 1.0f);
}

// ---------------------------------------------------------------------- cvt
__global__ void cvt_f32_to_f16_x4(const float* __restrict__ src,
                                  _Float16* __restrict__ dst, int n4) {
  int i = blockIdx.x * blockDim.x + threadIdx.x;
  if (i < n4) {
    v4f v = ((const v4f*)src)[i];
    v4h h;
    h[0] = (_Float16)v[0]; h[1] = (_Float16)v[1];
    h[2] = (_Float16)v[2]; h[3] = (_Float16)v[3];
    ((v4h*)dst)[i] = h;
  }
}

__global__ void zero_state(_Float16* __restrict__ h0, float* __restrict__ c0) {
  int i = blockIdx.x * blockDim.x + threadIdx.x;
  if (i < B_ * D_) { h0[i] = (_Float16)0.0f; c0[i] = 0.0f; }
}

// --------------------------------------------------------------- projection
// grid = (M/256, D/64, 3), block = 128 (4 waves). Each wave: 64x64 macro-tile
// (4x4 WMMA tiles, 16 accumulators). Output stored [T][B][D] in f16.
__global__ __launch_bounds__(128)
void proj_gemm(const _Float16* __restrict__ X16,   // [M_, D_]
               const _Float16* __restrict__ W16,   // [3][D_][D_]  (Wi,Wc,Wo)
               const float* __restrict__ bi,
               const float* __restrict__ bc,
               const float* __restrict__ bo,
               _Float16* __restrict__ XP) {        // [3][T][B][D]
  const int w    = blockIdx.z;
  const int lane = threadIdx.x & 31;
  const int wv   = threadIdx.x >> 5;
  const int lr   = lane & 15;
  const int hi   = lane >> 4;
  const int m0   = (blockIdx.x * 4 + wv) * 64;
  const int n0   = blockIdx.y * 64;

  const _Float16* W = W16 + (size_t)w * D_ * D_;
  v8f acc[4][4] = {};

  for (int k0 = 0; k0 < D_; k0 += 32) {
    v16h a[4], b[4];
#pragma unroll
    for (int i = 0; i < 4; ++i)
      a[i] = load_a_frag(X16, m0 + 16 * i + lr, k0, hi, D_);
#pragma unroll
    for (int j = 0; j < 4; ++j)
      b[j] = load_b_frag(W, n0 + 16 * j + lr, k0, hi, D_);
#pragma unroll
    for (int i = 0; i < 4; ++i)
#pragma unroll
      for (int j = 0; j < 4; ++j)
        acc[i][j] = wmma16(a[i], b[j], acc[i][j]);
  }

  const float* bptr = (w == 0) ? bi : (w == 1) ? bc : bo;
  float bv[4];
#pragma unroll
  for (int j = 0; j < 4; ++j) bv[j] = bptr[n0 + 16 * j + lr];

  _Float16* out = XP + (size_t)w * TBD;
#pragma unroll
  for (int i = 0; i < 4; ++i) {
#pragma unroll
    for (int r = 0; r < 8; ++r) {
      int m = m0 + 16 * i + r + hi * 8;      // flat row = b*T + t
      int t = m & (T_ - 1);
      int bb = m >> 9;                        // / T_
      _Float16* po = out + ((size_t)t * B_ + bb) * D_ + n0 + lr;
#pragma unroll
      for (int j = 0; j < 4; ++j)
        po[16 * j] = (_Float16)(acc[i][j][r] + bv[j]);
    }
  }
}

// --------------------------------------------------------------- recurrence
// grid = (32), block = 128 (4 waves). Block owns columns [n0, n0+16) of all
// four gates; waves split the batch (wv*16..). Per K chunk the 5KB U slab is
// async-copied into LDS (double buffered) and shared by all 4 waves.
__global__ __launch_bounds__(128)
void lstm_step(const _Float16* __restrict__ hprev,  // [B_][D_] f16
               const _Float16* __restrict__ U16,    // [4][D_][D_] (Ui,Uf,Uc,Uo)
               const _Float16* __restrict__ XP,     // [3][T][B][D] f16
               const float* __restrict__ bUf,
               const float* __restrict__ bUc,
               const float* __restrict__ bUo,
               float* __restrict__ cstate,          // [B_][D_] in/out
               _Float16* __restrict__ hnext,        // [B_][D_] f16
               float* __restrict__ Hout,            // [B_][T_][D_]
               float* __restrict__ Cout,            // [B_][T_][D_]
               int t) {
  __shared__ __attribute__((aligned(16))) char smem[2 * CHUNK_BYTES_];

  const int tid  = threadIdx.x;
  const int lane = tid & 31;
  const int wv   = tid >> 5;
  const int lr   = lane & 15;
  const int hi   = lane >> 4;
  const int m0   = wv * 16;
  const int n0   = blockIdx.x * 16;

  // Issue the async copy of one 32-wide K chunk of all 4 U slabs into LDS.
  // 256 b128 transfers per chunk; 2 per thread; 2 async instructions per wave.
  auto issue_chunk = [&](int k0, int bufsel) {
    const unsigned lbase =
        (unsigned)(unsigned long long)(smem + bufsel * CHUNK_BYTES_);
#pragma unroll
    for (int half = 0; half < 2; ++half) {
      const int e   = tid + half * 128;     // 0..255
      const int g   = e >> 6;               // gate
      const int row = (e >> 2) & 15;        // row within n-tile
      const int seg = e & 3;                // 16B segment within 64B of K
      const _Float16* gp =
          U16 + (size_t)g * D_ * D_ + (size_t)(n0 + row) * D_ + k0 + seg * 8;
      const unsigned lp = lbase + g * GSLAB_ + row * ROWSTRIDE_ + seg * 16;
      asm volatile("global_load_async_to_lds_b128 %0, %1, off"
                   :: "v"(lp), "v"(gp)
                   : "memory");
    }
  };

  v8f ai = {}, af = {}, ac = {}, ao = {};

  issue_chunk(0, 0);
  for (int kc = 0; kc < NKCH_; ++kc) {
    // Own chunk-kc copies done; barrier => everyone's copies landed AND every
    // wave has finished reading buffer (kc+1)&1 (consumed at iteration kc-1).
    asm volatile("s_wait_asynccnt 0x0" ::: "memory");
    __syncthreads();
    if (kc + 1 < NKCH_) issue_chunk((kc + 1) * 32, (kc + 1) & 1);

    const char* slab = smem + (kc & 1) * CHUNK_BYTES_;
    v16h a  = load_a_frag(hprev, m0 + lr, kc * 32, hi, D_);
    v16h b0 = lds_b_frag(slab, 0, lr, hi);
    v16h b1 = lds_b_frag(slab, 1, lr, hi);
    v16h b2 = lds_b_frag(slab, 2, lr, hi);
    v16h b3 = lds_b_frag(slab, 3, lr, hi);
    ai = wmma16(a, b0, ai);
    af = wmma16(a, b1, af);
    ac = wmma16(a, b2, ac);
    ao = wmma16(a, b3, ao);
  }

  const size_t tslice = (size_t)t * B_ * D_;
  const int n = n0 + lr;
  const float bf_n = bUf[n];
  const float bc_n = bUc[n];
  const float bo_n = bUo[n];

#pragma unroll
  for (int r = 0; r < 8; ++r) {
    const int bb = m0 + r + hi * 8;           // batch index
    const size_t sidx = (size_t)bb * D_ + n;  // state index
    const size_t pidx = tslice + sidx;        // preact slice index

    const float xi = (float)XP[pidx];
    const float xc = (float)XP[TBD + pidx];
    const float xo = (float)XP[2 * TBD + pidx];
    const float cp = cstate[sidx];

    const float ig = sigm_(xi + ai[r]);                 // i: no recurrent bias
    const float fg = sigm_(xi + af[r] + bf_n);          // f reuses xi (quirk)
    const float cn = fg * cp + ig * tanh_(xc + ac[r] + bc_n);
    const float og = sigm_(xo + ao[r] + bo_n);
    const float hn = og * tanh_(cn);

    cstate[sidx] = cn;
    hnext[sidx]  = (_Float16)hn;
    const size_t oidx = (size_t)bb * T_ * D_ + (size_t)t * D_ + n;
    Hout[oidx] = hn;
    Cout[oidx] = cn;
  }
}

// ------------------------------------------------------------------- launch
extern "C" void kernel_launch(void* const* d_in, const int* in_sizes, int n_in,
                              void* d_out, int out_size, void* d_ws, size_t ws_size,
                              hipStream_t stream) {
  const float* X   = (const float*)d_in[0];
  const float* Wi  = (const float*)d_in[1];
  const float* bi  = (const float*)d_in[2];
  // d_in[3] = Wf, d_in[4] = bf : dead in the reference (Xf is unused)
  const float* Wc  = (const float*)d_in[5];
  const float* bc  = (const float*)d_in[6];
  const float* Wo  = (const float*)d_in[7];
  const float* bo  = (const float*)d_in[8];
  const float* Ui  = (const float*)d_in[9];
  const float* Uf  = (const float*)d_in[10];
  const float* bUf = (const float*)d_in[11];
  const float* Uc  = (const float*)d_in[12];
  const float* bUc = (const float*)d_in[13];
  const float* Uo  = (const float*)d_in[14];
  const float* bUo = (const float*)d_in[15];

  char* ws = (char*)d_ws;
  size_t off = 0;
  _Float16* X16 = (_Float16*)(ws + off); off += (size_t)M_ * D_ * 2;      // 32 MB
  _Float16* W16 = (_Float16*)(ws + off); off += (size_t)3 * D_ * D_ * 2;  // 1.5 MB
  _Float16* U16 = (_Float16*)(ws + off); off += (size_t)4 * D_ * D_ * 2;  // 2 MB
  _Float16* XP  = (_Float16*)(ws + off); off += 3 * TBD * 2;              // 96 MB
  float*    CS  = (float*)   (ws + off); off += (size_t)B_ * D_ * 4;
  _Float16* H2  = (_Float16*)(ws + off); off += (size_t)2 * B_ * D_ * 2;

  float* Hout = (float*)d_out;
  float* Cout = Hout + (size_t)B_ * T_ * D_;

  // Phase 0: conversions + state init
  {
    const int nX4 = (M_ * D_) / 4;
    cvt_f32_to_f16_x4<<<nX4 / 256, 256, 0, stream>>>(X, X16, nX4);
    const int nW4 = (D_ * D_) / 4;
    cvt_f32_to_f16_x4<<<nW4 / 256, 256, 0, stream>>>(Wi, W16,                 nW4);
    cvt_f32_to_f16_x4<<<nW4 / 256, 256, 0, stream>>>(Wc, W16 + 1 * D_ * D_,   nW4);
    cvt_f32_to_f16_x4<<<nW4 / 256, 256, 0, stream>>>(Wo, W16 + 2 * D_ * D_,   nW4);
    cvt_f32_to_f16_x4<<<nW4 / 256, 256, 0, stream>>>(Ui, U16,                 nW4);
    cvt_f32_to_f16_x4<<<nW4 / 256, 256, 0, stream>>>(Uf, U16 + 1 * D_ * D_,   nW4);
    cvt_f32_to_f16_x4<<<nW4 / 256, 256, 0, stream>>>(Uc, U16 + 2 * D_ * D_,   nW4);
    cvt_f32_to_f16_x4<<<nW4 / 256, 256, 0, stream>>>(Uo, U16 + 3 * D_ * D_,   nW4);
    zero_state<<<(B_ * D_) / 256, 256, 0, stream>>>(H2, CS);
  }

  // Phase 1: Xi/Xc/Xo = X*W^T + b, stored [T,B,D]
  {
    dim3 grid(M_ / 256, D_ / 64, 3);
    proj_gemm<<<grid, 128, 0, stream>>>(X16, W16, bi, bc, bo, XP);
  }

  // Phase 2: 512 sequential steps (double-buffered h)
  for (int t = 0; t < T_; ++t) {
    _Float16* hprev = H2 + (size_t)(t & 1) * B_ * D_;
    _Float16* hnext = H2 + (size_t)((t + 1) & 1) * B_ * D_;
    lstm_step<<<32, 128, 0, stream>>>(hprev, U16, XP, bUf, bUc, bUo,
                                      CS, hnext, Hout, Cout, t);
  }
}
